// SelfAttention_558345748853
// MI455X (gfx1250) — matrix operand order repaired and checked by reading, not compiled, not run on hardware
//
#include <hip/hip_runtime.h>

// ---------------- types ----------------
typedef __bf16 bf16;
typedef __attribute__((ext_vector_type(16))) __bf16 v16bf;
typedef __attribute__((ext_vector_type(8)))  __bf16 v8bf;
typedef __attribute__((ext_vector_type(8)))  float   v8f;

#define DIM 2048
#define SEQ 2048
#define NH  32
#define NKV 8
#define HD  64
#define KVD (NKV*HD)   // 512

// fp32 -> bf16 round-to-nearest-even (bit-exact, no reliance on __bf16 arith)
__device__ __forceinline__ bf16 f2bf(float f) {
  unsigned u = __builtin_bit_cast(unsigned, f);
  u += 0x7FFFu + ((u >> 16) & 1u);
  unsigned short h = (unsigned short)(u >> 16);
  return __builtin_bit_cast(bf16, h);
}

__device__ __forceinline__ v16bf cat8(v8bf lo, v8bf hi) {
  return __builtin_shufflevector(lo, hi, 0,1,2,3,4,5,6,7,8,9,10,11,12,13,14,15);
}

__device__ __forceinline__ v16bf ld_frag(const bf16* p) {   // two contiguous 16B loads
  return cat8(*(const v8bf*)(p), *(const v8bf*)(p + 8));
}

__device__ __forceinline__ v8f wmma_bf16(v16bf a, v16bf b, v8f c) {
  return __builtin_amdgcn_wmma_f32_16x16x32_bf16(false, a, false, b, (short)0, c,
                                                 false, false);
}

// ---------------- fp32 -> bf16 convert (layout-preserving) ----------------
__global__ void cvt_f32_bf16(const float* __restrict__ in, bf16* __restrict__ out, int n) {
  int i = blockIdx.x * blockDim.x + threadIdx.x;
  if (i < n) out[i] = f2bf(in[i]);
}

// ---------------- fp32 [R][Cn] -> bf16 transposed [Cn][R], LDS-tiled ----------------
__global__ __launch_bounds__(256) void cvt_transpose(const float* __restrict__ in,
                                                     bf16* __restrict__ out,
                                                     int R, int Cn) {
  __shared__ float tile[32][33];
  const int c0 = blockIdx.x * 32;
  const int r0 = blockIdx.y * 32;
  const int tx = threadIdx.x & 31;
  const int ty = threadIdx.x >> 5;        // 0..7
  #pragma unroll
  for (int i = 0; i < 4; ++i)             // coalesced reads along input rows
    tile[ty + i * 8][tx] = in[(size_t)(r0 + ty + i * 8) * Cn + c0 + tx];
  __syncthreads();
  #pragma unroll
  for (int i = 0; i < 4; ++i)             // coalesced writes along output rows
    out[(size_t)(c0 + ty + i * 8) * R + r0 + tx] = f2bf(tile[tx][ty + i * 8]);
}

// ---------------- bf16 GEMM: C(f32)[M,N] = A[M,K] x Bt[N,K]^T ----------------
// block = 256 threads = 8 waves as 2(M) x 4(N); block tile 64x128; wave tile 32x32.
__global__ __launch_bounds__(256) void gemm_bf16(const bf16* __restrict__ A,
                                                 const bf16* __restrict__ Bt,
                                                 float* __restrict__ C,
                                                 int M, int N, int K) {
  const int lane = threadIdx.x & 31;
  const int wave = threadIdx.x >> 5;
  const int l15  = lane & 15;
  const int half = lane >> 4;
  const int bm = blockIdx.y * 64  + (wave >> 2) * 32;
  const int bn = blockIdx.x * 128 + (wave & 3) * 32;

  v8f acc[2][2] = {};

  // A frag: lane row = bm(+16)+l15; elements k = k0+half*16+e (contiguous)
  const bf16* Arow0 = A  + (bm + l15) * K + half * 16;
  const bf16* Arow1 = Arow0 + 16 * K;
  // B frag from Bt[n][k]: lane col = bn(+16)+l15; elements k = k0+half*16+e (contiguous)
  const bf16* Bcol0 = Bt + (bn + l15) * K + half * 16;
  const bf16* Bcol1 = Bcol0 + 16 * K;

  #pragma unroll 2
  for (int k0 = 0; k0 < K; k0 += 32) {
    __builtin_prefetch(Arow0 + k0 + 128, 0, 1);   // global_prefetch_b8
    __builtin_prefetch(Bcol0 + k0 + 128, 0, 1);
    v16bf a0 = cat8(*(const v8bf*)(Arow0 + k0), *(const v8bf*)(Arow0 + k0 + 8));
    v16bf a1 = cat8(*(const v8bf*)(Arow1 + k0), *(const v8bf*)(Arow1 + k0 + 8));
    v16bf b0 = ld_frag(Bcol0 + k0);
    v16bf b1 = ld_frag(Bcol1 + k0);
    acc[0][0] = wmma_bf16(a0, b0, acc[0][0]);
    acc[0][1] = wmma_bf16(a0, b1, acc[0][1]);
    acc[1][0] = wmma_bf16(a1, b0, acc[1][0]);
    acc[1][1] = wmma_bf16(a1, b1, acc[1][1]);
  }

  // C frag: lane col = l15, rows m = r + 8*half. 32-bit address math only.
  #pragma unroll
  for (int i = 0; i < 2; ++i)
    #pragma unroll
    for (int j = 0; j < 2; ++j) {
      int cbase = (bm + i * 16 + 8 * half) * N + bn + j * 16 + l15;
      #pragma unroll
      for (int r = 0; r < 8; ++r)
        C[cbase + r * N] = acc[i][j][r];
    }
}

// ---------------- RoPE: f32 in, bf16 out. One thread per (s,h,pair) ----------------
__global__ void rope_bf16(const float* __restrict__ T,
                          const float* __restrict__ Ct,
                          const float* __restrict__ St,
                          bf16* __restrict__ O, int n_heads) {
  int idx = blockIdx.x * blockDim.x + threadIdx.x;
  int total = SEQ * n_heads * (HD / 2);
  if (idx >= total) return;
  int p = idx & 31;                 // pair index within head (HD/2 = 32)
  int h = (idx >> 5) % n_heads;
  int s = idx / (n_heads * 32);
  float c = Ct[s * 32 + p], sn = St[s * 32 + p];
  int base = (s * n_heads + h) * HD + p * 2;
  float t0 = T[base], t1 = T[base + 1];
  O[base]     = f2bf(t0 * c - t1 * sn);
  O[base + 1] = f2bf(t0 * sn + t1 * c);
}

// ---------------- Flash attention: one wave per (head, 16-query tile) ----------------
// Q [S,2048] bf16 (post-RoPE), K [S,512] bf16 (post-RoPE), Vt [512,S] bf16 transposed.
__global__ __launch_bounds__(128) void attn_fwd(const bf16* __restrict__ Q,
                                                const bf16* __restrict__ Kb,
                                                const bf16* __restrict__ Vt,
                                                bf16* __restrict__ O) {
  __shared__ __attribute__((aligned(16))) bf16 lds_p[4][16 * 32]; // per-wave P tile
  const int lane = threadIdx.x & 31;
  const int wave = threadIdx.x >> 5;
  const int wid  = blockIdx.x * 4 + wave;
  const int h    = wid >> 7;        // 128 query tiles per head
  const int qb   = wid & 127;
  const int hk   = h >> 2;          // GQA: 4 query heads per KV head
  const int l15  = lane & 15;
  const int half = lane >> 4;

  // Q tile 16x64 as two A-fragments (d chunks 0..31, 32..63)
  const bf16* qrow = Q + (qb * 16 + l15) * DIM + h * HD + half * 16;
  v16bf qa0 = cat8(*(const v8bf*)(qrow),      *(const v8bf*)(qrow + 8));
  v16bf qa1 = cat8(*(const v8bf*)(qrow + 32), *(const v8bf*)(qrow + 40));

  float mi[8], li[8];
  #pragma unroll
  for (int r = 0; r < 8; ++r) { mi[r] = -__builtin_inff(); li[r] = 0.f; }
  v8f acc[4] = {};                  // O tile 16x64 (4 d-chunks of 16)

  bf16* pl = &lds_p[wave][0];
  const int nkb = ((qb * 16 + 15) >> 5) + 1;   // causal: key blocks of 32

  for (int kb = 0; kb < nkb; ++kb) {
    // ---- scores S = Q K^T : two 16x16 f32 fragments ----
    v8f s0 = {}, s1 = {};
    {
      // B-layout: lane -> key, elements -> contiguous head-dim (32B loads)
      const bf16* k0p = Kb + (kb * 32 + l15) * KVD + hk * HD + half * 16;
      const bf16* k1p = k0p + 16 * KVD;
      s0 = wmma_bf16(qa0, ld_frag(k0p),      s0);
      s0 = wmma_bf16(qa1, ld_frag(k0p + 32), s0);
      s1 = wmma_bf16(qa0, ld_frag(k1p),      s1);
      s1 = wmma_bf16(qa1, ld_frag(k1p + 32), s1);
    }

    // ---- mask + online softmax (per C-fragment row r -> m = r + 8*half) ----
    const int key0 = kb * 32 + l15;
    #pragma unroll
    for (int r = 0; r < 8; ++r) {
      const int q = qb * 16 + r + 8 * half;
      float v0 = (key0      <= q) ? s0[r] * 0.125f : -__builtin_inff();
      float v1 = (key0 + 16 <= q) ? s1[r] * 0.125f : -__builtin_inff();
      float mx = fmaxf(v0, v1);
      #pragma unroll
      for (int off = 1; off < 16; off <<= 1)
        mx = fmaxf(mx, __shfl_xor(mx, off, 32));     // stays within 16-lane group
      float nm   = fmaxf(mi[r], mx);                  // finite: key kb*32 <= q always
      float corr = __expf(mi[r] - nm);
      float p0   = __expf(v0 - nm);
      float p1   = __expf(v1 - nm);
      float ps = p0 + p1;
      #pragma unroll
      for (int off = 1; off < 16; off <<= 1)
        ps += __shfl_xor(ps, off, 32);
      li[r] = li[r] * corr + ps;
      mi[r] = nm;
      #pragma unroll
      for (int c2 = 0; c2 < 4; ++c2) acc[c2][r] *= corr;
      const int mrow = r + 8 * half;
      pl[mrow * 32 + l15]      = f2bf(p0);   // C-layout -> LDS (row-major 16x32)
      pl[mrow * 32 + 16 + l15] = f2bf(p1);
    }

    // ---- reload P as A-fragment (same wave: LDS ops in-order) ----
    const bf16* prow = pl + l15 * 32 + half * 16;
    v16bf pa = cat8(*(const v8bf*)(prow), *(const v8bf*)(prow + 8));

    // ---- O += P x V : B-frag from Vt[d][s] -> contiguous key loads ----
    #pragma unroll
    for (int c2 = 0; c2 < 4; ++c2) {
      const bf16* vrow = Vt + (hk * HD + c2 * 16 + l15) * SEQ + kb * 32 + half * 16;
      acc[c2] = wmma_bf16(pa, ld_frag(vrow), acc[c2]);
    }
  }

  // ---- normalize + store bf16 [q, h*64+d] ----
  #pragma unroll
  for (int c2 = 0; c2 < 4; ++c2) {
    int obase = (qb * 16 + 8 * half) * DIM + h * HD + c2 * 16 + l15;
    #pragma unroll
    for (int r = 0; r < 8; ++r)
      O[obase + r * DIM] = f2bf(acc[c2][r] / li[r]);
  }
}

// ---------------- host launcher ----------------
extern "C" void kernel_launch(void* const* d_in, const int* in_sizes, int n_in,
                              void* d_out, int out_size, void* d_ws, size_t ws_size,
                              hipStream_t stream) {
  (void)in_sizes; (void)n_in; (void)out_size; (void)ws_size;
  const float* x  = (const float*)d_in[0];
  const float* wq = (const float*)d_in[1];
  const float* wk = (const float*)d_in[2];
  const float* wv = (const float*)d_in[3];
  const float* wo = (const float*)d_in[4];
  const float* fc = (const float*)d_in[5];
  const float* fs = (const float*)d_in[6];
  float* out = (float*)d_out;

  char* ws = (char*)d_ws;
  const size_t MB = 1ull << 20;
  bf16*  xb  = (bf16*) (ws + 0);        //  8 MB  x bf16 [S,2048]
  bf16*  wqt = (bf16*) (ws +  8*MB);    //  8 MB  Wq^T [2048,2048]
  bf16*  wkt = (bf16*) (ws + 16*MB);    //  2 MB  Wk^T [512,2048]
  bf16*  wvt = (bf16*) (ws + 18*MB);    //  2 MB  Wv^T [512,2048]
  bf16*  wot = (bf16*) (ws + 20*MB);    //  8 MB  Wo^T [2048,2048]
  float* qf  = (float*)(ws + 28*MB);    // 16 MB  Q fp32 (pre-RoPE)
  float* kf  = (float*)(ws + 44*MB);    //  4 MB
  float* vf  = (float*)(ws + 48*MB);    //  4 MB
  bf16*  qbb = (bf16*) (ws + 52*MB);    //  8 MB  Q bf16 (post-RoPE) [S,2048]
  bf16*  kbb = (bf16*) (ws + 60*MB);    //  2 MB  K bf16 (post-RoPE) [S,512]
  bf16*  vt  = (bf16*) (ws + 62*MB);    //  2 MB  V^T bf16 [512,S]
  bf16*  ob  = (bf16*) (ws + 64*MB);    //  8 MB  attention output [S,2048]  (72 MB total)

  // x: keep layout; weights: transpose to [N][K] for contiguous B-fragments
  cvt_f32_bf16 <<<(SEQ*DIM + 255) / 256, 256, 0, stream>>>(x, xb, SEQ * DIM);
  cvt_transpose<<<dim3(DIM/32, DIM/32), 256, 0, stream>>>(wq, wqt, DIM, DIM);
  cvt_transpose<<<dim3(KVD/32, DIM/32), 256, 0, stream>>>(wk, wkt, DIM, KVD);
  cvt_transpose<<<dim3(KVD/32, DIM/32), 256, 0, stream>>>(wv, wvt, DIM, KVD);
  cvt_transpose<<<dim3(DIM/32, DIM/32), 256, 0, stream>>>(wo, wot, DIM, DIM);

  gemm_bf16<<<dim3(DIM/128, SEQ/64), 256, 0, stream>>>(xb, wqt, qf, SEQ, DIM, DIM);
  gemm_bf16<<<dim3(KVD/128, SEQ/64), 256, 0, stream>>>(xb, wkt, kf, SEQ, KVD, DIM);
  gemm_bf16<<<dim3(KVD/128, SEQ/64), 256, 0, stream>>>(xb, wvt, vf, SEQ, KVD, DIM);

  rope_bf16<<<(SEQ*NH *32 + 255) / 256, 256, 0, stream>>>(qf, fc, fs, qbb, NH);
  rope_bf16<<<(SEQ*NKV*32 + 255) / 256, 256, 0, stream>>>(kf, fc, fs, kbb, NKV);
  cvt_transpose<<<dim3(KVD/32, SEQ/32), 256, 0, stream>>>(vf, vt, SEQ, KVD); // V -> V^T

  attn_fwd<<<(NH * (SEQ/16)) / 4, 128, 0, stream>>>(qbb, kbb, vt, ob);

  gemm_bf16<<<dim3(DIM/128, SEQ/64), 256, 0, stream>>>(ob, wot, out, SEQ, DIM, DIM);
}
